// EquivariantNetwork_26147760898655
// MI455X (gfx1250) — matrix operand order repaired and checked by reading, not compiled, not run on hardware
//
#include <hip/hip_runtime.h>
#include <hip/hip_bf16.h>
#include <math.h>

// ---------------------------------------------------------------------------
// CDNA5 (gfx1250) EGNN kernel. Edge MLPs use V_WMMA_F32_16X16X32_BF16.
// bf16 operands staged in LDS row-major so A-fragments load as 2x ds_load_b128.
// ---------------------------------------------------------------------------

typedef __attribute__((ext_vector_type(16))) __bf16 v16bf;
typedef __attribute__((ext_vector_type(8)))  float  v8f;

#define BATCH 8
#define NNODE 192
#define HDIM  32
#define EDIM  66   // 2H + 2

typedef struct { uint4 lo, hi; } fragbits_t;  // 32 bytes == v16bf

__device__ __forceinline__ unsigned short bfbits(float f) {
  return __builtin_bit_cast(unsigned short, (__bf16)f);  // native cvt
}
__device__ __forceinline__ float sigmoidf_(float x) {
  return __builtin_amdgcn_rcpf(1.0f + __expf(-x));
}
__device__ __forceinline__ float siluf_(float x) { return x * sigmoidf_(x); }
__device__ __forceinline__ float softplusf_(float x) {
  return (x > 20.0f) ? x : log1pf(__expf(x));
}
__device__ __forceinline__ float fsqrt_(float x) {
  return __builtin_amdgcn_sqrtf(x);  // raw v_sqrt_f32, no IEEE refinement tail
}

__device__ __forceinline__ v8f wmma_bf16(v16bf a, v16bf b, v8f c) {
  return __builtin_amdgcn_wmma_f32_16x16x32_bf16(
      false, a, false, b, (short)0, c, false, false);
}

// A-fragment from a bf16 row-major row pointer (32 K values, 16B aligned).
// Element e holds K = ((e<8)? e : e+8) + half*8  ->  two contiguous 16B loads.
__device__ __forceinline__ v16bf afrag_row(const unsigned short* rowp, int half) {
  fragbits_t fb;
  fb.lo = *(const uint4*)(rowp + half * 8);
  fb.hi = *(const uint4*)(rowp + 16 + half * 8);
  return __builtin_bit_cast(v16bf, fb);
}

// B-fragment (K=32 x N=16 bf16) from W row-major [out_dim][in_dim]
// (lin computes x @ W^T, so B[k][n] = W[n][k]).
__device__ __forceinline__ v16bf load_bfrag(const float* __restrict__ W,
                                            int out_dim, int in_dim,
                                            int kbase, int ntile, int lane) {
  v16bf r;
  int n  = ntile * 16 + (lane & 15);
  int kb = kbase + ((lane >> 4) * 8);
#pragma unroll
  for (int e = 0; e < 16; e++) {
    int k = kb + ((e < 8) ? e : e + 8);
    float v = (n < out_dim && k < in_dim) ? W[n * in_dim + k] : 0.0f;
    r[e] = (__bf16)v;
  }
  return r;
}

// ---------------------------------------------------------------------------
// Front-end: gamma(t) MLP, pdf embedding, input embedding 12->32.
// ---------------------------------------------------------------------------
__device__ float block_reduce256(float v, float* s_red) {
  int tid = threadIdx.x;
  s_red[tid] = v;
  __syncthreads();
  for (int s = 128; s > 0; s >>= 1) {
    if (tid < s) s_red[tid] += s_red[tid + s];
    __syncthreads();
  }
  float r = s_red[0];
  __syncthreads();
  return r;
}

__global__ __launch_bounds__(256) void frontend_kernel(
    const float* __restrict__ xyz, const float* __restrict__ pdf,
    const float* __restrict__ tt,
    const float* __restrict__ emb_in_w, const float* __restrict__ emb_in_b,
    const float* __restrict__ pdf_w, const float* __restrict__ pdf_b,
    const float* __restrict__ g1w, const float* __restrict__ g1b,
    const float* __restrict__ g2w, const float* __restrict__ g2b,
    const float* __restrict__ g3w, const float* __restrict__ g3b,
    const float* __restrict__ gam0, const float* __restrict__ gam1,
    float* __restrict__ h0, float* __restrict__ xcur, float* __restrict__ x0) {
  __shared__ float s_red[256];
  __shared__ float s_temb;
  __shared__ float s_pemb[10];
  const int b = blockIdx.x;
  const int tid = threadIdx.x;

  float tn   = tt[b] * (1.0f / 1000.0f);
  float w11  = softplusf_(g1w[0]);
  float b11  = g1b[0];
  float l1_0 = b11, l1_1 = w11 + b11, l1_t = w11 * tn + b11;
  float p0 = 0.f, p1 = 0.f, pt = 0.f;
  for (int k = tid; k < 1024; k += 256) {
    float w2 = softplusf_(g2w[k]);
    float b2 = g2b[k];
    float w3 = softplusf_(g3w[k]);
    p0 += w3 * sigmoidf_(w2 * l1_0 + b2);
    p1 += w3 * sigmoidf_(w2 * l1_1 + b2);
    pt += w3 * sigmoidf_(w2 * l1_t + b2);
  }
  float s0 = block_reduce256(p0, s_red);
  float s1 = block_reduce256(p1, s_red);
  float st = block_reduce256(pt, s_red);
  if (tid == 0) {
    float b3  = g3b[0];
    float g0v = l1_0 + s0 + b3;
    float g1v = l1_1 + s1 + b3;
    float gtv = l1_t + st + b3;
    s_temb = gam0[0] + (gam1[0] - gam0[0]) * (gtv - g0v) *
                           __builtin_amdgcn_rcpf(g1v - g0v);
  }
  for (int o = 0; o < 10; o++) {
    float p = 0.f;
    for (int k = tid; k < 3000; k += 256)
      p += pdf[b * 3000 + k] * pdf_w[o * 3000 + k];
    float s = block_reduce256(p, s_red);
    if (tid == 0) s_pemb[o] = s + pdf_b[o];
  }
  __syncthreads();

  for (int idx = tid; idx < NNODE * HDIM; idx += 256) {
    int n = idx >> 5, f = idx & 31;
    float sp  = xyz[(b * NNODE + n) * 4 + 3];
    float acc = emb_in_b[f];
    acc += emb_in_w[f * 12 + 0] * sp;
    acc += emb_in_w[f * 12 + 1] * s_temb;
#pragma unroll
    for (int o = 0; o < 10; o++) acc += emb_in_w[f * 12 + 2 + o] * s_pemb[o];
    h0[(b * NNODE + n) * HDIM + f] = acc;
  }
  for (int idx = tid; idx < NNODE * 3; idx += 256) {
    int n = idx / 3, c = idx % 3;
    float v = xyz[(b * NNODE + n) * 4 + c];
    xcur[(b * NNODE + n) * 3 + c] = v;
    x0[(b * NNODE + n) * 3 + c]   = v;
  }
}

// ---------------------------------------------------------------------------
// One EGC layer. Grid: B*12 blocks (16 "i" nodes each), 256 threads (8 waves).
// ---------------------------------------------------------------------------
__global__ __launch_bounds__(256) void egc_layer_kernel(
    const float* __restrict__ h_in, const float* __restrict__ x_in,
    const float* __restrict__ x0g,
    float* __restrict__ h_out, float* __restrict__ x_out, int layer,
    const float* __restrict__ lnh_g, const float* __restrict__ lnh_b,
    const float* __restrict__ lnx_g, const float* __restrict__ lnx_b,
    const float* __restrict__ edg1_w, const float* __restrict__ edg1_b,
    const float* __restrict__ edg2_w, const float* __restrict__ edg2_b,
    const float* __restrict__ edgi_w, const float* __restrict__ edgi_b,
    const float* __restrict__ node1_w, const float* __restrict__ node1_b,
    const float* __restrict__ node2_w, const float* __restrict__ node2_b,
    const float* __restrict__ cor1_w, const float* __restrict__ cor1_b,
    const float* __restrict__ cor2_w, const float* __restrict__ cor2_b,
    const float* __restrict__ cor3_w, const float* __restrict__ cor3_b) {
  __shared__ float s_hn[NNODE * HDIM];                         // normed h (f32)
  __shared__ __align__(16) unsigned short s_hbf[NNODE * HDIM]; // normed h (bf16)
  __shared__ float s_xr[NNODE * 3];
  __shared__ float s_xn[NNODE * 3];
  __shared__ float s_x0[NNODE * 3];
  __shared__ float s_magg[16 * HDIM];
  __shared__ float s_xacc[16 * 3];
  __shared__ float s_eb1[HDIM], s_eb2[HDIM], s_cb1[HDIM], s_cb2[HDIM];
  __shared__ float s_eiw[HDIM], s_c3w[HDIM];
  __shared__ float s_eib, s_c3b;
  __shared__ __align__(16) unsigned short s_mscr[8][16 * HDIM]; // per-wave bf16 tile
  __shared__ __align__(16) float s_e[8][16];                    // per-wave gates
  __shared__ float s_hn1[16 * HDIM];

  const int b     = blockIdx.x / 12;
  const int itile = blockIdx.x % 12;
  const int ibase = itile * 16;
  const int tid   = threadIdx.x;

  // ---- Stage 1: LayerNorm h,x into LDS -----------------------------------
  if (tid < NNODE) {
    const float* hp = h_in + (b * NNODE + tid) * HDIM;
    float m = 0.f;
    for (int f = 0; f < HDIM; f++) m += hp[f];
    m *= (1.0f / HDIM);
    float v = 0.f;
    for (int f = 0; f < HDIM; f++) { float d = hp[f] - m; v += d * d; }
    v *= (1.0f / HDIM);
    float inv = rsqrtf(v + 1e-5f);
    for (int f = 0; f < HDIM; f++) {
      float z = (hp[f] - m) * inv * lnh_g[layer * HDIM + f] + lnh_b[layer * HDIM + f];
      s_hn[tid * HDIM + f]  = z;
      s_hbf[tid * HDIM + f] = bfbits(z);
    }
    const float* xp = x_in + (b * NNODE + tid) * 3;
    float xv0 = xp[0], xv1 = xp[1], xv2 = xp[2];
    s_xr[tid * 3 + 0] = xv0; s_xr[tid * 3 + 1] = xv1; s_xr[tid * 3 + 2] = xv2;
    float mx = (xv0 + xv1 + xv2) * (1.0f / 3.0f);
    float vx = ((xv0 - mx) * (xv0 - mx) + (xv1 - mx) * (xv1 - mx) +
                (xv2 - mx) * (xv2 - mx)) * (1.0f / 3.0f);
    float invx = rsqrtf(vx + 1e-5f);
    s_xn[tid * 3 + 0] = (xv0 - mx) * invx * lnx_g[layer * 3 + 0] + lnx_b[layer * 3 + 0];
    s_xn[tid * 3 + 1] = (xv1 - mx) * invx * lnx_g[layer * 3 + 1] + lnx_b[layer * 3 + 1];
    s_xn[tid * 3 + 2] = (xv2 - mx) * invx * lnx_g[layer * 3 + 2] + lnx_b[layer * 3 + 2];
    const float* x0p = x0g + (b * NNODE + tid) * 3;
    s_x0[tid * 3 + 0] = x0p[0]; s_x0[tid * 3 + 1] = x0p[1]; s_x0[tid * 3 + 2] = x0p[2];
  }
  for (int k = tid; k < 16 * HDIM; k += 256) s_magg[k] = 0.f;
  if (tid < 48) s_xacc[tid] = 0.f;
  if (tid < HDIM) {
    s_eb1[tid] = edg1_b[layer * HDIM + tid];
    s_eb2[tid] = edg2_b[layer * HDIM + tid];
    s_cb1[tid] = cor1_b[layer * HDIM + tid];
    s_cb2[tid] = cor2_b[layer * HDIM + tid];
    s_eiw[tid] = edgi_w[layer * HDIM + tid];
    s_c3w[tid] = cor3_w[layer * HDIM + tid];
  }
  if (tid == 0) { s_eib = edgi_b[layer]; s_c3b = cor3_b[layer]; }
  __syncthreads();

  // ---- Stage 2: weight B-fragments in registers --------------------------
  const int lane = tid & 31;
  const int wv   = tid >> 5;
  const int half = lane >> 4;
  const int rown = lane & 15;

  v16bf We1[3][2], Wc1[3][2], We2[2], Wc2[2];
#pragma unroll
  for (int c = 0; c < 3; c++)
#pragma unroll
    for (int n = 0; n < 2; n++) {
      We1[c][n] = load_bfrag(edg1_w + layer * HDIM * EDIM, HDIM, EDIM, c * 32, n, lane);
      Wc1[c][n] = load_bfrag(cor1_w + layer * HDIM * EDIM, HDIM, EDIM, c * 32, n, lane);
    }
#pragma unroll
  for (int n = 0; n < 2; n++) {
    We2[n] = load_bfrag(edg2_w + layer * HDIM * HDIM, HDIM, HDIM, 0, n, lane);
    Wc2[n] = load_bfrag(cor2_w + layer * HDIM * HDIM, HDIM, HDIM, 0, n, lane);
  }
  unsigned short* msc = &s_mscr[wv][0];

  // ---- Stage 3: edge-tile loop (16 edges per WMMA M-tile) ----------------
  for (int t = wv; t < 192; t += 8) {
    int i_loc = t / 12;
    int jt    = t - i_loc * 12;
    int i     = ibase + i_loc;
    int j     = jt * 16 + rown;  // this lane's edge-row target node

    float dx0 = s_xr[i * 3 + 0] - s_xr[j * 3 + 0];
    float dx1 = s_xr[i * 3 + 1] - s_xr[j * 3 + 1];
    float dx2 = s_xr[i * 3 + 2] - s_xr[j * 3 + 2];
    float d2  = dx0 * dx0 + dx1 * dx1 + dx2 * dx2;
    float e00 = s_x0[i * 3 + 0] - s_x0[j * 3 + 0];
    float e01 = s_x0[i * 3 + 1] - s_x0[j * 3 + 1];
    float e02 = s_x0[i * 3 + 2] - s_x0[j * 3 + 2];
    float dd0 = e00 * e00 + e01 * e01 + e02 * e02;
    float dist = fsqrt_(fmaxf(d2, 1e-12f));
    float d0   = fsqrt_(fmaxf(dd0, 1e-12f));

    // A fragments: h_i row (uniform), h_j row (per lane), [d2,d0,0...]
    v16bf a0 = afrag_row(s_hbf + i * HDIM, half);
    v16bf a1 = afrag_row(s_hbf + j * HDIM, half);
    v16bf a2 = {};
    a2[0] = (__bf16)((half == 0) ? d2 : 0.0f);  // K==0
    a2[1] = (__bf16)((half == 0) ? d0 : 0.0f);  // K==1

    // ---- edge branch: m = silu(edg2(silu(edg1(einp)))) ----
#pragma unroll
    for (int n = 0; n < 2; n++) {
      v8f acc = {};
      acc = wmma_bf16(a0, We1[0][n], acc);
      acc = wmma_bf16(a1, We1[1][n], acc);
      acc = wmma_bf16(a2, We1[2][n], acc);
      int col = n * 16 + rown;
      float bia = s_eb1[col];
#pragma unroll
      for (int v = 0; v < 8; v++) {
        float z = siluf_(acc[v] + bia);
        msc[(v + half * 8) * HDIM + col] = bfbits(z);
      }
    }
    v16bf am = afrag_row(msc + rown * HDIM, half);
    v8f m2v[2];
#pragma unroll
    for (int n = 0; n < 2; n++) {
      v8f acc = {};
      acc = wmma_bf16(am, We2[n], acc);
      int col = n * 16 + rown;
      float bia = s_eb2[col];
#pragma unroll
      for (int v = 0; v < 8; v++) {
        float z = siluf_(acc[v] + bia);
        m2v[n][v] = z;
        msc[(v + half * 8) * HDIM + col] = bfbits(z);
      }
    }
    // edge gate: e = sigmoid(edgi . m) (packed u32 reads), diagonal masked
    {
      float ev = s_eib;
      const unsigned* m32 = (const unsigned*)(msc + rown * HDIM);
#pragma unroll
      for (int k2 = 0; k2 < HDIM / 2; k2++) {
        unsigned u = m32[k2];
        ev += s_eiw[2 * k2]     * __builtin_bit_cast(float, u << 16);
        ev += s_eiw[2 * k2 + 1] * __builtin_bit_cast(float, u & 0xFFFF0000u);
      }
      ev = sigmoidf_(ev);
      if (j == i) ev = 0.f;
      if (half == 0) s_e[wv][rown] = ev;
    }
    // m_agg[i] += sum_rows e * m  (lane owns one column, 8 rows)
    {
      const float4* ep = (const float4*)&s_e[wv][half * 8];
      float4 ea = ep[0], eb = ep[1];
#pragma unroll
      for (int n = 0; n < 2; n++) {
        float part = ea.x * m2v[n][0] + ea.y * m2v[n][1] + ea.z * m2v[n][2] +
                     ea.w * m2v[n][3] + eb.x * m2v[n][4] + eb.y * m2v[n][5] +
                     eb.z * m2v[n][6] + eb.w * m2v[n][7];
        atomicAdd(&s_magg[i_loc * HDIM + n * 16 + rown], part);
      }
    }

    // ---- coordinate branch: cw = cor3(silu(cor2(silu(cor1(einp))))) ----
#pragma unroll
    for (int n = 0; n < 2; n++) {
      v8f acc = {};
      acc = wmma_bf16(a0, Wc1[0][n], acc);
      acc = wmma_bf16(a1, Wc1[1][n], acc);
      acc = wmma_bf16(a2, Wc1[2][n], acc);
      int col = n * 16 + rown;
      float bia = s_cb1[col];
#pragma unroll
      for (int v = 0; v < 8; v++) {
        float z = siluf_(acc[v] + bia);
        msc[(v + half * 8) * HDIM + col] = bfbits(z);
      }
    }
    v16bf ac = afrag_row(msc + rown * HDIM, half);
#pragma unroll
    for (int n = 0; n < 2; n++) {
      v8f acc = {};
      acc = wmma_bf16(ac, Wc2[n], acc);
      int col = n * 16 + rown;
      float bia = s_cb2[col];
#pragma unroll
      for (int v = 0; v < 8; v++) {
        float z = siluf_(acc[v] + bia);
        msc[(v + half * 8) * HDIM + col] = bfbits(z);
      }
    }
    {
      float cw = s_c3b;
      const unsigned* m32 = (const unsigned*)(msc + rown * HDIM);
#pragma unroll
      for (int k2 = 0; k2 < HDIM / 2; k2++) {
        unsigned u = m32[k2];
        cw += s_c3w[2 * k2]     * __builtin_bit_cast(float, u << 16);
        cw += s_c3w[2 * k2 + 1] * __builtin_bit_cast(float, u & 0xFFFF0000u);
      }
      if (half == 0) {
        float sc = cw * __builtin_amdgcn_rcpf(dist + 1.0f);
        atomicAdd(&s_xacc[i_loc * 3 + 0], sc * dx0);
        atomicAdd(&s_xacc[i_loc * 3 + 1], sc * dx1);
        atomicAdd(&s_xacc[i_loc * 3 + 2], sc * dx2);
      }
    }
  }
  __syncthreads();

  // ---- Stage 4: node MLP + residuals -------------------------------------
  {
    int ni = tid >> 4;
    int f0 = tid & 15;
    int i  = ibase + ni;
    for (int f = f0; f < HDIM; f += 16) {
      const float* w1 = node1_w + (layer * HDIM + f) * (2 * HDIM);
      float acc = node1_b[layer * HDIM + f];
#pragma unroll
      for (int k = 0; k < HDIM; k++) acc += w1[k] * s_hn[i * HDIM + k];
#pragma unroll
      for (int k = 0; k < HDIM; k++) acc += w1[HDIM + k] * s_magg[ni * HDIM + k];
      s_hn1[ni * HDIM + f] = siluf_(acc);
    }
  }
  __syncthreads();
  {
    int ni = tid >> 4;
    int f0 = tid & 15;
    int i  = ibase + ni;
    for (int f = f0; f < HDIM; f += 16) {
      const float* w2 = node2_w + (layer * HDIM + f) * HDIM;
      float acc = node2_b[layer * HDIM + f];
#pragma unroll
      for (int k = 0; k < HDIM; k++) acc += w2[k] * s_hn1[ni * HDIM + k];
      h_out[(b * NNODE + i) * HDIM + f] = s_hn[i * HDIM + f] + acc;
    }
    if (tid < 48) {
      int nn = tid / 3, c = tid % 3, ii = ibase + nn;
      x_out[(b * NNODE + ii) * 3 + c] = s_xn[ii * 3 + c] + s_xacc[nn * 3 + c];
    }
  }
}

// ---------------------------------------------------------------------------
__global__ __launch_bounds__(256) void finalize_kernel(
    const float* __restrict__ hf, const float* __restrict__ xf,
    const float* __restrict__ x0,
    const float* __restrict__ xow, const float* __restrict__ xob,
    const float* __restrict__ eow, const float* __restrict__ eob,
    float* __restrict__ out) {
  int idx = blockIdx.x * blockDim.x + threadIdx.x;
  if (idx >= BATCH * NNODE) return;
  const float* xp = xf + idx * 3;
#pragma unroll
  for (int r = 0; r < 3; r++) {
    float acc = xob[r];
#pragma unroll
    for (int c = 0; c < 3; c++) acc += xow[r * 3 + c] * xp[c];
    out[idx * 4 + r] = acc - x0[idx * 3 + r];
  }
  float acc = eob[0];
  const float* hp = hf + idx * HDIM;
#pragma unroll
  for (int f = 0; f < HDIM; f++) acc += eow[f] * hp[f];
  out[idx * 4 + 3] = acc;
}

// ---------------------------------------------------------------------------
extern "C" void kernel_launch(void* const* d_in, const int* in_sizes, int n_in,
                              void* d_out, int out_size, void* d_ws, size_t ws_size,
                              hipStream_t stream) {
  (void)in_sizes; (void)n_in; (void)out_size; (void)ws_size;
  const float* xyz      = (const float*)d_in[0];
  const float* pdf      = (const float*)d_in[1];
  const float* tt       = (const float*)d_in[2];
  const float* emb_in_w = (const float*)d_in[3];
  const float* emb_in_b = (const float*)d_in[4];
  const float* emb_out_w= (const float*)d_in[5];
  const float* emb_out_b= (const float*)d_in[6];
  const float* x_out_w  = (const float*)d_in[7];
  const float* x_out_b  = (const float*)d_in[8];
  const float* pdf_w    = (const float*)d_in[9];
  const float* pdf_b    = (const float*)d_in[10];
  const float* g1w      = (const float*)d_in[11];
  const float* g1b      = (const float*)d_in[12];
  const float* g2w      = (const float*)d_in[13];
  const float* g2b      = (const float*)d_in[14];
  const float* g3w      = (const float*)d_in[15];
  const float* g3b      = (const float*)d_in[16];
  const float* gam0     = (const float*)d_in[17];
  const float* gam1     = (const float*)d_in[18];
  const float* lnh_g    = (const float*)d_in[19];
  const float* lnh_b    = (const float*)d_in[20];
  const float* lnx_g    = (const float*)d_in[21];
  const float* lnx_b    = (const float*)d_in[22];
  const float* edg1_w   = (const float*)d_in[23];
  const float* edg1_b   = (const float*)d_in[24];
  const float* edg2_w   = (const float*)d_in[25];
  const float* edg2_b   = (const float*)d_in[26];
  const float* edgi_w   = (const float*)d_in[27];
  const float* edgi_b   = (const float*)d_in[28];
  const float* node1_w  = (const float*)d_in[29];
  const float* node1_b  = (const float*)d_in[30];
  const float* node2_w  = (const float*)d_in[31];
  const float* node2_b  = (const float*)d_in[32];
  const float* cor1_w   = (const float*)d_in[33];
  const float* cor1_b   = (const float*)d_in[34];
  const float* cor2_w   = (const float*)d_in[35];
  const float* cor2_b   = (const float*)d_in[36];
  const float* cor3_w   = (const float*)d_in[37];
  const float* cor3_b   = (const float*)d_in[38];

  float* ws = (float*)d_ws;
  const int XSZ = BATCH * NNODE * 3;
  const int HSZ = BATCH * NNODE * HDIM;
  float* x0 = ws;
  float* xA = x0 + XSZ;
  float* xB = xA + XSZ;
  float* hA = xB + XSZ;
  float* hB = hA + HSZ;

  frontend_kernel<<<BATCH, 256, 0, stream>>>(
      xyz, pdf, tt, emb_in_w, emb_in_b, pdf_w, pdf_b,
      g1w, g1b, g2w, g2b, g3w, g3b, gam0, gam1, hA, xA, x0);

  float* hc = hA; float* xc = xA;
  float* hn = hB; float* xn = xB;
  for (int l = 0; l < 4; l++) {
    egc_layer_kernel<<<BATCH * 12, 256, 0, stream>>>(
        hc, xc, x0, hn, xn, l,
        lnh_g, lnh_b, lnx_g, lnx_b,
        edg1_w, edg1_b, edg2_w, edg2_b, edgi_w, edgi_b,
        node1_w, node1_b, node2_w, node2_b,
        cor1_w, cor1_b, cor2_w, cor2_b, cor3_w, cor3_b);
    float* th = hc; hc = hn; hn = th;
    float* tx = xc; xc = xn; xn = tx;
  }

  finalize_kernel<<<(BATCH * NNODE + 255) / 256, 256, 0, stream>>>(
      hc, xc, x0, x_out_w, x_out_b, emb_out_w, emb_out_b, (float*)d_out);
}
